// RNNSharedNet_52003464020023
// MI455X (gfx1250) — matrix operand order repaired and checked by reading, not compiled, not run on hardware
//
#include <hip/hip_runtime.h>

#define T_ 35
#define B_ 64
#define V_ 32000
#define E_ 1024
#define H_ 1024
#define STEPS_ 8

typedef __attribute__((ext_vector_type(2)))  float  v2f;
typedef __attribute__((ext_vector_type(8)))  float  v8f;
typedef __attribute__((ext_vector_type(16))) __bf16 v16bf;

__device__ __forceinline__ unsigned short f32_to_bf16(float f) {
  unsigned int u = __builtin_bit_cast(unsigned int, f);
  u += 0x7fffu + ((u >> 16) & 1u);       // round-to-nearest-even
  return (unsigned short)(u >> 16);
}
__device__ __forceinline__ float sigmoidf_(float x) { return 1.0f / (1.0f + __expf(-x)); }
__device__ __forceinline__ float act_apply(int act, float x) {
  switch (act) {
    case 0: return sigmoidf_(x);         // sigmoid
    case 1: return fmaxf(x, 0.0f);       // relu
    case 2: return tanhf(x);             // tanh
    default: return x;                   // identity
  }
}

// Pack two 16-byte chunks (8 bf16 each) into a WMMA v16bf operand.
struct U4x2 { uint4 lo, hi; };
__device__ __forceinline__ v16bf pack16(uint4 lo, uint4 hi) {
  U4x2 t; t.lo = lo; t.hi = hi;
  return __builtin_bit_cast(v16bf, t);
}

// Async global -> LDS copy (gfx1250, tracked by ASYNCcnt). lds_addr is the
// 32-bit LDS byte address (== low 32 bits of the flat shared-memory pointer,
// per the LDS aperture mapping addr[31:0] -> LDS offset).
__device__ __forceinline__ void async_copy_b128(unsigned lds_addr,
                                                const unsigned short* gptr) {
  asm volatile("global_load_async_to_lds_b128 %0, %1, off"
               :: "v"(lds_addr), "v"(gptr) : "memory");
}
__device__ __forceinline__ void wait_async0() {
  asm volatile("s_wait_asynccnt 0" ::: "memory");
}

// ---------------------------------------------------------------- embeddings
__global__ void embw_to_bf16_kernel(const float* __restrict__ emb_w,
                                    unsigned short* __restrict__ out, long n) {
  long i = (long)blockIdx.x * blockDim.x + threadIdx.x;
  if (i < n) out[i] = f32_to_bf16(emb_w[i]);
}

__global__ void gather_embed_kernel(const int* __restrict__ tokens,
                                    const float* __restrict__ emb_w,
                                    float* __restrict__ emb) {
  int row = blockIdx.x;                          // t*B + b
  int tok = tokens[row];
  const float4* src = (const float4*)(emb_w + (size_t)tok * E_);
  float4* dst = (float4*)(emb + (size_t)row * E_);
  dst[threadIdx.x] = src[threadIdx.x];           // blockDim.x == 256 -> 1024 floats
}

// ------------------------------------------------------------- DARTS cell GEMM
// One wave per 16x16 output tile pair: accumulates the "c" tile at column n0
// and the "hh" tile at column n0+1024 with V_WMMA_F32_16X16X4_F32, then fuses
// the gating epilogue  s = sp + sigmoid(c) * (act(hh) - sp).
__device__ __forceinline__ void wmma_f32_pair(const float* __restrict__ Ar,
                                              const float* __restrict__ Wr0,
                                              const float* __restrict__ Wr1,
                                              v8f& dc, v8f& dh) {
  v2f a, bc, bh;
  a.x  = Ar[0];     a.y  = Ar[1];
  bc.x = Wr0[0];    bc.y = Wr1[0];               // columns [n0, n0+16)    -> c
  bh.x = Wr0[1024]; bh.y = Wr1[1024];            // columns [n0+1024, ...) -> hh
  dc = __builtin_amdgcn_wmma_f32_16x16x4_f32(false, a, false, bc, (short)0, dc, false, false);
  dh = __builtin_amdgcn_wmma_f32_16x16x4_f32(false, a, false, bh, (short)0, dh, false, false);
}

__global__ __launch_bounds__(32)
void cell_gemm_kernel(const float* __restrict__ A0,  // [B_,1024]  x (or sp)
                      const float* __restrict__ A1,  // [B_,1024]  h_prev (K=2048) or null
                      const float* __restrict__ W,   // [K, 2H]
                      const float* __restrict__ SP,  // [B_, H] state being gated
                      float* __restrict__ Sout,      // [B_, H]
                      float* __restrict__ acc,       // [B_, H] mean accumulator
                      int act_id, int acc_mode)      // acc_mode 0: acc=0 ; 1: acc+=s
{
  const int lane = threadIdx.x & 31;
  const int half = lane >> 4;                    // K sub-pair select (ISA A16x4 layout)
  const int lm   = lane & 15;
  const int m0   = blockIdx.y * 16;
  const int n0   = blockIdx.x * 16;
  const int row  = m0 + lm;

  v8f dc = {};
  v8f dh = {};

  // Phase 1: K = 0..1023 from A0 (x for GEMM0, sp for steps).
  {
    const float* Ab = A0 + (size_t)row * 1024;
    for (int k0 = 0; k0 < 1024; k0 += 4) {
      const int ka = k0 + 2 * half;              // lanes 0-15: K=k0,k0+1 ; 16-31: k0+2,k0+3
      wmma_f32_pair(Ab + ka,
                    W + (size_t)ka * 2048 + n0 + lm,
                    W + (size_t)(ka + 1) * 2048 + n0 + lm, dc, dh);
    }
  }
  // Phase 2: K = 1024..2047 from A1 (h_prev), only for the concat GEMM.
  if (A1) {
    const float* Ab = A1 + (size_t)row * 1024;
    const float* Wb = W + (size_t)1024 * 2048;
    for (int k0 = 0; k0 < 1024; k0 += 4) {
      const int ka = k0 + 2 * half;
      wmma_f32_pair(Ab + ka,
                    Wb + (size_t)ka * 2048 + n0 + lm,
                    Wb + (size_t)(ka + 1) * 2048 + n0 + lm, dc, dh);
    }
  }

#pragma unroll
  for (int r = 0; r < 8; ++r) {
    const int m   = m0 + r + 8 * half;           // C/D layout: VGPR r = M=r (lo) / M=r+8 (hi)
    const int idx = m * H_ + n0 + lm;
    float sp = SP[idx];
    float s  = sp + sigmoidf_(dc[r]) * (act_apply(act_id, dh[r]) - sp);
    Sout[idx] = s;
    if (acc_mode == 0) acc[idx] = 0.0f; else acc[idx] += s;
  }
}

__global__ void cell_finish_kernel(const float* __restrict__ acc,
                                   float* __restrict__ raw_t,
                                   float* __restrict__ droped_t,
                                   unsigned short* __restrict__ rawbf_t,
                                   float* __restrict__ next_h) {
  int i = blockIdx.x * blockDim.x + threadIdx.x;
  if (i < B_ * H_) {
    float v = acc[i] * 0.125f;                   // mean over the 8 step states
    raw_t[i]    = v;
    droped_t[i] = v;                             // eval-mode dropout is identity
    rawbf_t[i]  = f32_to_bf16(v);
    if (next_h) next_h[i] = v;
  }
}

// ----------------------------------------------------------- tied decoder GEMM
// logits[m, v] = sum_k outs[m,k] * emb_w[v,k]   (M=2240, N=32000, K=1024)
// bf16 inputs, f32 accumulate: V_WMMA_F32_16X16X32_BF16.
// 4 waves / block share one M tile: its A panel (16 x 1024 bf16 = 32 KB) is
// staged in LDS once via async global->LDS copies, then fed to WMMA with
// ds_load_b128 reads; the streamed B panel stays in global with prefetch.
__global__ __launch_bounds__(128)
void decoder_gemm_kernel(const unsigned short* __restrict__ Abf,  // [T*B, 1024] bf16
                         const unsigned short* __restrict__ Wbf,  // [V, 1024]   bf16
                         float* __restrict__ logits) {
  __shared__ __align__(16) unsigned short As[16 * 1024];          // 32 KB A panel

  const int tid  = threadIdx.x;
  const int wave = tid >> 5;
  const int lane = tid & 31;
  const int half = lane >> 4;
  const int lm   = lane & 15;
  const int m0   = blockIdx.y * 16;
  const int n0   = (blockIdx.x * 4 + wave) * 16;

  // ---- async stage of the A panel: 2048 x 16B chunks, 16 per thread ----
  {
    const unsigned as_base = (unsigned)(unsigned long long)(const void*)As;
    const unsigned short* g = Abf + (size_t)m0 * 1024;
#pragma unroll
    for (int i = 0; i < 16; ++i) {
      const int idx = i * 128 + tid;             // linear 16B chunk id (row-major panel)
      async_copy_b128(as_base + (unsigned)idx * 16, g + (size_t)idx * 8);
    }
  }
  wait_async0();
  __syncthreads();

  const uint4* Asv = (const uint4*)As;                            // 8 bf16 per uint4
  const uint4* Bv  = (const uint4*)(Wbf + (size_t)(n0 + lm) * 1024);

  v8f d = {};
  for (int k0 = 0; k0 < 1024; k0 += 32) {
    const int kc = k0 >> 3;                      // uint4 index of this K chunk
    __builtin_prefetch(Bv + kc + 32, 0, 1);      // global_prefetch_b8 on streamed B
    // A: lo lanes K 0-7/16-23, hi lanes 8-15/24-31 (16-bit A 16x32 layout)
    uint4 a_lo = Asv[lm * 128 + kc + half];
    uint4 a_hi = Asv[lm * 128 + kc + half + 2];
    // B: lo lanes K 0-7/16-23, hi lanes 8-15/24-31 (same K order as A)
    uint4 b_lo = Bv[kc + half];
    uint4 b_hi = Bv[kc + half + 2];
    d = __builtin_amdgcn_wmma_f32_16x16x32_bf16(false, pack16(a_lo, a_hi),
                                                false, pack16(b_lo, b_hi),
                                                (short)0, d, false, false);
  }

#pragma unroll
  for (int r = 0; r < 8; ++r) {
    const int m = m0 + r + 8 * half;
    logits[(size_t)m * V_ + n0 + lm] = d[r];
  }
}

// ---------------------------------------------------------------------- driver
extern "C" void kernel_launch(void* const* d_in, const int* in_sizes, int n_in,
                              void* d_out, int out_size, void* d_ws, size_t ws_size,
                              hipStream_t stream) {
  (void)in_sizes; (void)n_in; (void)out_size; (void)ws_size;
  const int*   tokens  = (const int*)d_in[0];
  const float* hiddens = (const float*)d_in[1];
  const float* emb_w   = (const float*)d_in[2];
  const float* W0      = (const float*)d_in[3];
  const float* Ws      = (const float*)d_in[4];

  float* out    = (float*)d_out;
  float* logits = out;                                     // [T,B,V]
  float* raw    = logits + (size_t)T_ * B_ * V_;           // [T,B,H]
  float* droped = raw    + (size_t)T_ * B_ * H_;           // [T,B,H]
  float* next_h = droped + (size_t)T_ * B_ * H_;           // [1,B,H]

  char* w = (char*)d_ws;
  unsigned short* embw_bf = (unsigned short*)w; w += (size_t)V_ * E_ * 2;       // 65.5 MB
  float*          emb     = (float*)w;          w += (size_t)T_ * B_ * E_ * 4;  // 9.2 MB
  unsigned short* raw_bf  = (unsigned short*)w; w += (size_t)T_ * B_ * H_ * 2;  // 4.6 MB
  float* states[STEPS_ + 1];
  for (int i = 0; i <= STEPS_; ++i) { states[i] = (float*)w; w += (size_t)B_ * H_ * 4; }
  float* accb = (float*)w;                      w += (size_t)B_ * H_ * 4;

  // emb_w -> bf16 (decoder B matrix)
  {
    long n = (long)V_ * E_;
    embw_to_bf16_kernel<<<dim3((unsigned)((n + 255) / 256)), dim3(256), 0, stream>>>(
        emb_w, embw_bf, n);
  }
  // token gather
  gather_embed_kernel<<<dim3(T_ * B_), dim3(256), 0, stream>>>(tokens, emb_w, emb);

  // DARTS_V1 genotype
  static const int g_frm[STEPS_] = {0, 1, 1, 1, 2, 5, 3, 5};
  static const int g_act[STEPS_] = {0, 1, 1, 3, 2, 0, 2, 1};  // sig,relu,relu,id,tanh,sig,tanh,relu

  const dim3 cell_grid(H_ / 16, B_ / 16), cell_blk(32);
  for (int t = 0; t < T_; ++t) {
    const float* h_prev = (t == 0) ? hiddens : (raw + (size_t)(t - 1) * B_ * H_);
    // s0 = h_prev + sigmoid(c) * (tanh(h) - h_prev), ch = [x|h_prev] @ W0
    cell_gemm_kernel<<<cell_grid, cell_blk, 0, stream>>>(
        emb + (size_t)t * B_ * E_, h_prev, W0, h_prev, states[0], accb,
        /*act=tanh*/2, /*acc=0*/0);
    for (int i = 0; i < STEPS_; ++i) {
      const float* sp = states[g_frm[i]];
      cell_gemm_kernel<<<cell_grid, cell_blk, 0, stream>>>(
          sp, nullptr, Ws + (size_t)i * H_ * 2 * H_, sp, states[i + 1], accb,
          g_act[i], /*acc+=*/1);
    }
    cell_finish_kernel<<<dim3((B_ * H_ + 255) / 256), dim3(256), 0, stream>>>(
        accb, raw + (size_t)t * B_ * H_, droped + (size_t)t * B_ * H_,
        raw_bf + (size_t)t * B_ * H_, (t == T_ - 1) ? next_h : nullptr);
  }

  // logits = outs @ emb_w^T   (grid: 2000 N-tiles / 4 per block, 140 M-tiles)
  decoder_gemm_kernel<<<dim3(V_ / 64, (T_ * B_) / 16), dim3(128), 0, stream>>>(
      raw_bf, embw_bf, logits);
}
